// BasicDNC_850403525342
// MI455X (gfx1250) — matrix operand order repaired and checked by reading, not compiled, not run on hardware
//
#include <hip/hip_runtime.h>
#include <math.h>

// ---------------------------------------------------------------------------
// BasicDNC forward for MI455X (gfx1250, wave32, WMMA bf16).
// fp32 params -> bf16 operands (one-time cast/transpose into ws), all GEMMs
// via v_wmma_f32_16x16x32_bf16 (fp32 accum) with 32x64 per-wave tiles,
// flash-style attention (online softmax, no T*T materialization), chunked
// top-8 external memory read, tied-embedding logits GEMM into d_out.
// ---------------------------------------------------------------------------

typedef __attribute__((ext_vector_type(16))) __bf16 v16bf;
typedef __attribute__((ext_vector_type(8)))  float  v8f;

union V16 { v16bf v; uint4 q[2]; };
union V8  { v8f   v; float f[8]; };

__device__ __forceinline__ v8f wmma_bf16(v16bf a, v16bf b, v8f c) {
  // D = A(16x32 bf16) * B(32x16 bf16) + C(16x16 f32)
  return __builtin_amdgcn_wmma_f32_16x16x32_bf16(
      /*neg_a=*/false, a, /*neg_b=*/false, b,
      /*c_mod=*/(short)0, c, /*reuse_a=*/false, /*reuse_b=*/false);
}

#define DNC_D     512
#define DNC_FF    2048
#define DNC_T     2048
#define DNC_B     2
#define DNC_H     8
#define DNC_DH    64
#define DNC_M     4096       // B*T tokens
#define DNC_V     32000
#define DNC_SLOTS 32768
#define DNC_CH    2048       // slot chunk for scores
#define DNC_NCH   16
#define DNC_L     4

// ---------------------------------------------------------------------------
// Elementwise helpers
// ---------------------------------------------------------------------------
__global__ void dnc_cast_bf16(const float* __restrict__ in,
                              __bf16* __restrict__ out) {
  size_t i = (size_t)blockIdx.x * 256 + threadIdx.x;
  out[i] = (__bf16)in[i];
}

// in: fp32 [K,N] row-major  ->  out: bf16 [N,K] row-major (Bt operand form)
__global__ void dnc_transpose_cast(const float* __restrict__ in,
                                   __bf16* __restrict__ out,
                                   int K, int N) {
  size_t e = (size_t)blockIdx.x * 256 + threadIdx.x;
  int k = (int)(e / N), n = (int)(e % N);
  out[(size_t)n * K + k] = (__bf16)in[e];
}

// x = tok_embed[ids] + pos_embed[t]; also bf16 mirror
__global__ void dnc_embed(const int* __restrict__ ids,
                          const float* __restrict__ tok,
                          const float* __restrict__ pos,
                          float* __restrict__ x, __bf16* __restrict__ x_bf) {
  int row = blockIdx.x;                 // 0..4095 (b*T + t)
  int t   = row & (DNC_T - 1);
  int id  = ids[row];
  for (int d = threadIdx.x; d < DNC_D; d += 256) {
    float v = tok[(size_t)id * DNC_D + d] + pos[(size_t)t * DNC_D + d];
    x[(size_t)row * DNC_D + d]    = v;
    x_bf[(size_t)row * DNC_D + d] = (__bf16)v;
  }
}

// RMSNorm over D=512, writes fp32 + bf16 mirror
__global__ void dnc_rmsnorm(const float* __restrict__ in,
                            const float* __restrict__ w,
                            float* __restrict__ out,
                            __bf16* __restrict__ out_bf) {
  int row = blockIdx.x, t = threadIdx.x;
  const float* r = in + (size_t)row * DNC_D;
  float v0 = r[t], v1 = r[t + 256];
  __shared__ float red[256];
  red[t] = v0 * v0 + v1 * v1;
  __syncthreads();
  for (int s = 128; s > 0; s >>= 1) {
    if (t < s) red[t] += red[t + s];
    __syncthreads();
  }
  __shared__ float inv_s;
  if (t == 0) inv_s = rsqrtf(red[0] * (1.0f / DNC_D) + 1e-8f);
  __syncthreads();
  float inv = inv_s;
  float o0 = v0 * inv * w[t], o1 = v1 * inv * w[t + 256];
  out[(size_t)row * DNC_D + t]          = o0;
  out[(size_t)row * DNC_D + t + 256]    = o1;
  out_bf[(size_t)row * DNC_D + t]       = (__bf16)o0;
  out_bf[(size_t)row * DNC_D + t + 256] = (__bf16)o1;
}

// q,k,v fp32 [M, D] -> Qh/Kh bf16 [B,H,T,dh], Vt bf16 [B,H,dh,T]
__global__ void dnc_pack_qkv(const float* __restrict__ q,
                             const float* __restrict__ k,
                             const float* __restrict__ v,
                             __bf16* __restrict__ Qh, __bf16* __restrict__ Kh,
                             __bf16* __restrict__ Vt) {
  size_t e = (size_t)blockIdx.x * 256 + threadIdx.x;   // e < M*D
  int row = (int)(e >> 9), d = (int)(e & 511);
  int b = row >> 11, t = row & (DNC_T - 1);
  int h = d >> 6, c = d & 63;
  size_t bh = (size_t)(b * DNC_H + h);
  Qh[(bh * DNC_T + t) * DNC_DH + c] = (__bf16)q[e];
  Kh[(bh * DNC_T + t) * DNC_DH + c] = (__bf16)k[e];
  Vt[(bh * DNC_DH + c) * DNC_T + t] = (__bf16)v[e];
}

// ---------------------------------------------------------------------------
// Generic WMMA GEMM: C[M,N] = act(A[M,K]bf16 * Bt[N,K]bf16 + bias) + residual
// One wave computes a 32x64 tile (2 M-subtiles x 4 N-subtiles); each B
// fragment is reused against 2 A fragments, halving B-side L2 traffic.
// Block = 4 waves = 32x256 of C. grid = (N/256, M/32), block = 128.
// ---------------------------------------------------------------------------
__global__ void dnc_gemm_bf16(const __bf16* __restrict__ A,
                              const __bf16* __restrict__ Bt,
                              const float* __restrict__ bias,
                              const float* __restrict__ residual,
                              float* __restrict__ Cf,
                              __bf16* __restrict__ Cbf,
                              int M, int N, int K, int act) {
  int lane = threadIdx.x & 31;
  int wave = threadIdx.x >> 5;
  int n0 = (blockIdx.x * 4 + wave) * 64;
  int m0 = blockIdx.y * 32;
  int mA    = lane & 15;          // A-operand row within 16-tile
  int kb    = (lane >> 4) * 8;    // A-row prefetch offset {0,8}
  int nlane = lane & 15;          // B-operand column
  int khalf = (lane >> 4) * 16;   // B-operand K half {0,16}

  const __bf16* arow0 = A + (size_t)(m0 + mA) * K;
  const __bf16* arow1 = A + (size_t)(m0 + 16 + mA) * K;

  V8 acc[8];   // [mt*4 + nt]
#pragma unroll
  for (int j = 0; j < 8; ++j)
#pragma unroll
    for (int i = 0; i < 8; ++i) acc[j].f[i] = 0.0f;

  for (int k = 0; k < K; k += 32) {
    // prefetch next K-slice of the A rows (global_prefetch_b8 path)
    if (k + 32 < K) {
      __builtin_prefetch(arow0 + k + 32 + kb, 0, 0);
      __builtin_prefetch(arow1 + k + 32 + kb, 0, 0);
    }
    V16 a0, a1;
    a0.q[0] = *reinterpret_cast<const uint4*>(arow0 + k + khalf);
    a0.q[1] = *reinterpret_cast<const uint4*>(arow0 + k + khalf + 8);
    a1.q[0] = *reinterpret_cast<const uint4*>(arow1 + k + khalf);
    a1.q[1] = *reinterpret_cast<const uint4*>(arow1 + k + khalf + 8);
#pragma unroll
    for (int nt = 0; nt < 4; ++nt) {
      const __bf16* brow = Bt + (size_t)(n0 + nt * 16 + nlane) * K + k + khalf;
      V16 b;
      b.q[0] = *reinterpret_cast<const uint4*>(brow);
      b.q[1] = *reinterpret_cast<const uint4*>(brow + 8);
      acc[nt].v     = wmma_bf16(a0.v, b.v, acc[nt].v);
      acc[4 + nt].v = wmma_bf16(a1.v, b.v, acc[4 + nt].v);
    }
  }

  int hi = lane >> 4;
#pragma unroll
  for (int mt = 0; mt < 2; ++mt) {
#pragma unroll
    for (int nt = 0; nt < 4; ++nt) {
      int col = n0 + nt * 16 + nlane;
      float bv = bias ? bias[col] : 0.0f;
#pragma unroll
      for (int r = 0; r < 8; ++r) {
        int row = m0 + mt * 16 + r + 8 * hi;
        float v = acc[mt * 4 + nt].f[r] + bv;
        if (act == 1) {  // tanh-approx GELU (jax.nn.gelu default)
          v = 0.5f * v * (1.0f + tanhf(0.7978845608028654f *
                                       (v + 0.044715f * v * v * v)));
        }
        size_t off = (size_t)row * N + col;
        if (residual) v += residual[off];
        if (Cf)  Cf[off]  = v;
        if (Cbf) Cbf[off] = (__bf16)v;
      }
    }
  }
}

// ---------------------------------------------------------------------------
// Flash attention: one wave per (bh, 16-row q tile). Full (non-causal) attn.
// Q*K^T via WMMA, online softmax, P*V via WMMA with an LDS re-layout of P
// (C-fragment -> A-fragment). Writes ao directly as bf16 [M, D].
// grid = (T/16, B*H), block = 32.
// ---------------------------------------------------------------------------
__global__ void dnc_attention(const __bf16* __restrict__ Qh,
                              const __bf16* __restrict__ Kh,
                              const __bf16* __restrict__ Vt,
                              __bf16* __restrict__ ao_bf) {
  const float scale = 0.125f;  // dh^-0.5, dh = 64
  int lane = threadIdx.x & 31;
  int m0 = blockIdx.x * 16;
  int bh = blockIdx.y;
  int b = bh >> 3, h = bh & 7;

  int nlane = lane & 15;
  int khalf = (lane >> 4) * 16;
  int kb    = (lane >> 4) * 8;
  int hi    = lane >> 4;

  const __bf16* Qb = Qh + (size_t)bh * DNC_T * DNC_DH;
  const __bf16* Kb = Kh + (size_t)bh * DNC_T * DNC_DH;
  const __bf16* Vb = Vt + (size_t)bh * DNC_DH * DNC_T;

  // Q A-fragments for head-dim K=0..31 and 32..63
  const __bf16* arow = Qb + (size_t)(m0 + (lane & 15)) * DNC_DH;
  V16 aq0, aq1;
  aq0.q[0] = *reinterpret_cast<const uint4*>(arow + khalf);
  aq0.q[1] = *reinterpret_cast<const uint4*>(arow + khalf + 8);
  aq1.q[0] = *reinterpret_cast<const uint4*>(arow + 32 + khalf);
  aq1.q[1] = *reinterpret_cast<const uint4*>(arow + 32 + khalf + 8);

  V8 o[4];
#pragma unroll
  for (int j = 0; j < 4; ++j)
#pragma unroll
    for (int i = 0; i < 8; ++i) o[j].f[i] = 0.0f;

  float rm[8], rl[8];
#pragma unroll
  for (int r = 0; r < 8; ++r) { rm[r] = -1e30f; rl[r] = 0.0f; }

  __shared__ __attribute__((aligned(16))) __bf16 plds[16 * 32];

  for (int kt = 0; kt < DNC_T; kt += 32) {
    // ---- S = Q * K^T for 32 key columns (two 16x16 tiles) ----
    V8 s0, s1;
#pragma unroll
    for (int i = 0; i < 8; ++i) { s0.f[i] = 0.0f; s1.f[i] = 0.0f; }
    {
      const __bf16* kr0 = Kb + (size_t)(kt + nlane) * DNC_DH + khalf;
      const __bf16* kr1 = Kb + (size_t)(kt + 16 + nlane) * DNC_DH + khalf;
      V16 bk;
      bk.q[0] = *reinterpret_cast<const uint4*>(kr0);
      bk.q[1] = *reinterpret_cast<const uint4*>(kr0 + 8);
      s0.v = wmma_bf16(aq0.v, bk.v, s0.v);
      bk.q[0] = *reinterpret_cast<const uint4*>(kr0 + 32);
      bk.q[1] = *reinterpret_cast<const uint4*>(kr0 + 40);
      s0.v = wmma_bf16(aq1.v, bk.v, s0.v);
      bk.q[0] = *reinterpret_cast<const uint4*>(kr1);
      bk.q[1] = *reinterpret_cast<const uint4*>(kr1 + 8);
      s1.v = wmma_bf16(aq0.v, bk.v, s1.v);
      bk.q[0] = *reinterpret_cast<const uint4*>(kr1 + 32);
      bk.q[1] = *reinterpret_cast<const uint4*>(kr1 + 40);
      s1.v = wmma_bf16(aq1.v, bk.v, s1.v);
    }

    // ---- online softmax update; stage P (bf16) into LDS ----
#pragma unroll
    for (int r = 0; r < 8; ++r) {
      float a0 = s0.f[r] * scale, a1 = s1.f[r] * scale;
      float tm = fmaxf(a0, a1);
      tm = fmaxf(tm, __shfl_xor(tm, 1, 32));
      tm = fmaxf(tm, __shfl_xor(tm, 2, 32));
      tm = fmaxf(tm, __shfl_xor(tm, 4, 32));
      tm = fmaxf(tm, __shfl_xor(tm, 8, 32));
      float nm = fmaxf(rm[r], tm);
      float f  = __expf(rm[r] - nm);
      float p0 = __expf(a0 - nm), p1 = __expf(a1 - nm);
      float ps = p0 + p1;
      ps += __shfl_xor(ps, 1, 32);
      ps += __shfl_xor(ps, 2, 32);
      ps += __shfl_xor(ps, 4, 32);
      ps += __shfl_xor(ps, 8, 32);
      rl[r] = rl[r] * f + ps;
      rm[r] = nm;
      o[0].f[r] *= f; o[1].f[r] *= f; o[2].f[r] *= f; o[3].f[r] *= f;
      int row = r + 8 * hi;
      plds[row * 32 + nlane]      = (__bf16)p0;
      plds[row * 32 + 16 + nlane] = (__bf16)p1;
    }
    __syncthreads();

    // ---- P A-fragment from LDS (16x32) ----
    V16 pA;
    {
      const __bf16* pp = &plds[(lane & 15) * 32 + khalf];
      pA.q[0] = *reinterpret_cast<const uint4*>(pp);
      pA.q[1] = *reinterpret_cast<const uint4*>(pp + 8);
    }

    // ---- O += P * V (4 x 16-col dh tiles) ----
#pragma unroll
    for (int nt = 0; nt < 4; ++nt) {
      const __bf16* vrow = Vb + (size_t)(nt * 16 + nlane) * DNC_T + kt + khalf;
      V16 bv;
      bv.q[0] = *reinterpret_cast<const uint4*>(vrow);
      bv.q[1] = *reinterpret_cast<const uint4*>(vrow + 8);
      o[nt].v = wmma_bf16(pA.v, bv.v, o[nt].v);
    }
    __syncthreads();
  }

  // ---- normalize and write ao bf16 [M, D] ----
#pragma unroll
  for (int r = 0; r < 8; ++r) {
    float inv = 1.0f / rl[r];
    int row = m0 + r + 8 * hi;
    size_t base = (size_t)(b * DNC_T + row) * DNC_D + h * DNC_DH;
#pragma unroll
    for (int nt = 0; nt < 4; ++nt)
      ao_bf[base + nt * 16 + nlane] = (__bf16)(o[nt].f[r] * inv);
  }
}

// ---------------------------------------------------------------------------
// Top-8 merge over one 2048-slot score chunk. One block per token row.
// Applies D^-0.5 scale + salience here (GEMM writes raw dot products).
// ---------------------------------------------------------------------------
__global__ void dnc_topk_merge(const float* __restrict__ chunk,
                               const float* __restrict__ salience,
                               float* __restrict__ top_s,
                               int* __restrict__ top_i,
                               int chunk_off, int is_first) {
  const float kscale = 0.04419417382415922f;  // 512^-0.5
  int row = blockIdx.x, t = threadIdx.x;
  float lv[8]; int li[8];
#pragma unroll
  for (int u = 0; u < 8; ++u) {
    int j = u * 256 + t;
    lv[u] = chunk[(size_t)row * DNC_CH + j] * kscale + salience[chunk_off + j];
    li[u] = chunk_off + j;
  }
  __shared__ float sv[256];
  __shared__ int   si[256];
  float wv8[8]; int wi8[8];
  for (int round = 0; round < 8; ++round) {
    float bvv = -1e30f; int bii = -1;
#pragma unroll
    for (int u = 0; u < 8; ++u)
      if (lv[u] > bvv) { bvv = lv[u]; bii = li[u]; }
    sv[t] = bvv; si[t] = bii;
    __syncthreads();
    for (int s = 128; s > 0; s >>= 1) {
      if (t < s) {
        if (sv[t + s] > sv[t] ||
            (sv[t + s] == sv[t] && si[t + s] < si[t])) {
          sv[t] = sv[t + s]; si[t] = si[t + s];
        }
      }
      __syncthreads();
    }
    float wv = sv[0]; int wi = si[0];
    __syncthreads();
    wv8[round] = wv; wi8[round] = wi;
#pragma unroll
    for (int u = 0; u < 8; ++u)
      if (li[u] == wi) lv[u] = -1e30f;
  }
  if (t == 0) {
    float cs[16]; int ci[16];
    for (int k = 0; k < 8; ++k) {
      cs[k] = is_first ? -1e30f : top_s[row * 8 + k];
      ci[k] = is_first ? -1     : top_i[row * 8 + k];
      cs[8 + k] = wv8[k]; ci[8 + k] = wi8[k];
    }
    for (int o2 = 0; o2 < 8; ++o2) {
      int best = 0;
      for (int k = 1; k < 16; ++k) if (cs[k] > cs[best]) best = k;
      top_s[row * 8 + o2] = cs[best];
      top_i[row * 8 + o2] = ci[best];
      cs[best] = -1e30f;
    }
  }
}

// softmax over top-8, gather mem_V rows, weighted sum -> read (bf16)
__global__ void dnc_memread(const float* __restrict__ top_s,
                            const int* __restrict__ top_i,
                            const float* __restrict__ mem_V,
                            __bf16* __restrict__ read_bf) {
  int row = blockIdx.x, t = threadIdx.x;
  __shared__ float w8[8];
  __shared__ int   i8[8];
  if (t == 0) {
    float s[8], mx = -1e30f;
    for (int k = 0; k < 8; ++k) { s[k] = top_s[row * 8 + k]; mx = fmaxf(mx, s[k]); }
    float sum = 0.0f, e[8];
    for (int k = 0; k < 8; ++k) { e[k] = __expf(s[k] - mx); sum += e[k]; }
    float inv = 1.0f / sum;
    for (int k = 0; k < 8; ++k) { w8[k] = e[k] * inv; i8[k] = top_i[row * 8 + k]; }
  }
  __syncthreads();
  for (int d = t; d < DNC_D; d += 256) {
    float acc = 0.0f;
#pragma unroll
    for (int k = 0; k < 8; ++k)
      acc += w8[k] * mem_V[(size_t)i8[k] * DNC_D + d];
    read_bf[(size_t)row * DNC_D + d] = (__bf16)acc;
  }
}

// ---------------------------------------------------------------------------
// Host driver
// ---------------------------------------------------------------------------
extern "C" void kernel_launch(void* const* d_in, const int* in_sizes, int n_in,
                              void* d_out, int out_size, void* d_ws, size_t ws_size,
                              hipStream_t stream) {
  (void)in_sizes; (void)n_in; (void)out_size; (void)ws_size;

  const int*   input_ids = (const int*)  d_in[0];
  const float* tok_embed = (const float*)d_in[1];
  const float* pos_embed = (const float*)d_in[2];
  const float* blk_wq    = (const float*)d_in[3];
  const float* blk_wk    = (const float*)d_in[4];
  const float* blk_wv    = (const float*)d_in[5];
  const float* blk_wo    = (const float*)d_in[6];
  const float* blk_bo    = (const float*)d_in[7];
  const float* blk_norm1 = (const float*)d_in[8];
  const float* blk_ffw1  = (const float*)d_in[9];
  const float* blk_ffb1  = (const float*)d_in[10];
  const float* blk_ffw2  = (const float*)d_in[11];
  const float* blk_ffb2  = (const float*)d_in[12];
  const float* blk_norm2 = (const float*)d_in[13];
  const float* mem_K     = (const float*)d_in[14];
  const float* mem_V     = (const float*)d_in[15];
  const float* salience  = (const float*)d_in[16];
  const float* wq_mem    = (const float*)d_in[17];
  const float* bq_mem    = (const float*)d_in[18];
  const float* w_read    = (const float*)d_in[19];
  const float* b_read    = (const float*)d_in[20];
  const float* norm_out  = (const float*)d_in[21];

  // ---- carve workspace (~200 MB) ----
  char* p = (char*)d_ws;
  auto carve = [&](size_t bytes) -> void* {
    void* r = (void*)p;
    p += (bytes + 255) & ~(size_t)255;
    return r;
  };
  float*  x       = (float*) carve((size_t)DNC_M * DNC_D * 4);
  __bf16* x_bf    = (__bf16*)carve((size_t)DNC_M * DNC_D * 2);
  float*  y       = (float*) carve((size_t)DNC_M * DNC_D * 4);
  float*  qf      = (float*) carve((size_t)DNC_M * DNC_D * 4);
  float*  kf      = (float*) carve((size_t)DNC_M * DNC_D * 4);
  float*  vf      = (float*) carve((size_t)DNC_M * DNC_D * 4);
  __bf16* Qh      = (__bf16*)carve((size_t)DNC_M * DNC_D * 2);
  __bf16* Kh      = (__bf16*)carve((size_t)DNC_M * DNC_D * 2);
  __bf16* Vt      = (__bf16*)carve((size_t)DNC_M * DNC_D * 2);
  __bf16* ao_bf   = (__bf16*)carve((size_t)DNC_M * DNC_D * 2);
  __bf16* h_bf    = (__bf16*)carve((size_t)DNC_M * DNC_FF * 2);
  __bf16* wt      = (__bf16*)carve((size_t)DNC_D * DNC_FF * 2);
  __bf16* mq_bf   = (__bf16*)carve((size_t)DNC_M * DNC_D * 2);
  __bf16* memK_bf = (__bf16*)carve((size_t)DNC_SLOTS * DNC_D * 2);
  __bf16* tokE_bf = (__bf16*)carve((size_t)DNC_V * DNC_D * 2);
  float*  chunk   = (float*) carve((size_t)DNC_M * DNC_CH * 4);
  float*  top_s   = (float*) carve((size_t)DNC_M * 8 * 4);
  int*    top_i   = (int*)   carve((size_t)DNC_M * 8 * 4);
  __bf16* read_bf = (__bf16*)carve((size_t)DNC_M * DNC_D * 2);

  auto gemm = [&](const __bf16* A, const __bf16* Bt, const float* bias,
                  const float* res, float* Cf, __bf16* Cbf,
                  int M, int N, int K, int act) {
    dim3 g(N / 256, M / 32);
    dnc_gemm_bf16<<<g, 128, 0, stream>>>(A, Bt, bias, res, Cf, Cbf, M, N, K, act);
  };
  auto tcast = [&](const float* in, __bf16* out, int K, int N) {
    dnc_transpose_cast<<<(unsigned)(((size_t)K * N) / 256), 256, 0, stream>>>(in, out, K, N);
  };

  // one-time casts of [N,K]-shaped params
  dnc_cast_bf16<<<(unsigned)(((size_t)DNC_V * DNC_D) / 256), 256, 0, stream>>>(tok_embed, tokE_bf);
  dnc_cast_bf16<<<(unsigned)(((size_t)DNC_SLOTS * DNC_D) / 256), 256, 0, stream>>>(mem_K, memK_bf);

  // embedding
  dnc_embed<<<DNC_M, 256, 0, stream>>>(input_ids, tok_embed, pos_embed, x, x_bf);

  // transformer blocks
  for (int l = 0; l < DNC_L; ++l) {
    const float* wq = blk_wq + (size_t)l * DNC_D * DNC_D;
    const float* wk = blk_wk + (size_t)l * DNC_D * DNC_D;
    const float* wv = blk_wv + (size_t)l * DNC_D * DNC_D;
    const float* wo = blk_wo + (size_t)l * DNC_D * DNC_D;
    const float* bo = blk_bo + (size_t)l * DNC_D;
    const float* n1 = blk_norm1 + (size_t)l * DNC_D;
    const float* f1 = blk_ffw1 + (size_t)l * DNC_D * DNC_FF;
    const float* fb1 = blk_ffb1 + (size_t)l * DNC_FF;
    const float* f2 = blk_ffw2 + (size_t)l * DNC_FF * DNC_D;
    const float* fb2 = blk_ffb2 + (size_t)l * DNC_D;
    const float* n2 = blk_norm2 + (size_t)l * DNC_D;

    tcast(wq, wt, DNC_D, DNC_D);
    gemm(x_bf, wt, nullptr, nullptr, qf, nullptr, DNC_M, DNC_D, DNC_D, 0);
    tcast(wk, wt, DNC_D, DNC_D);
    gemm(x_bf, wt, nullptr, nullptr, kf, nullptr, DNC_M, DNC_D, DNC_D, 0);
    tcast(wv, wt, DNC_D, DNC_D);
    gemm(x_bf, wt, nullptr, nullptr, vf, nullptr, DNC_M, DNC_D, DNC_D, 0);

    dnc_pack_qkv<<<(unsigned)(((size_t)DNC_M * DNC_D) / 256), 256, 0, stream>>>(
        qf, kf, vf, Qh, Kh, Vt);

    dnc_attention<<<dim3(DNC_T / 16, DNC_B * DNC_H), 32, 0, stream>>>(Qh, Kh, Vt, ao_bf);

    tcast(wo, wt, DNC_D, DNC_D);
    gemm(ao_bf, wt, bo, /*res=*/x, y, nullptr, DNC_M, DNC_D, DNC_D, 0);
    dnc_rmsnorm<<<DNC_M, 256, 0, stream>>>(y, n1, x, x_bf);

    tcast(f1, wt, DNC_D, DNC_FF);
    gemm(x_bf, wt, fb1, nullptr, nullptr, h_bf, DNC_M, DNC_FF, DNC_D, /*gelu=*/1);
    tcast(f2, wt, DNC_FF, DNC_D);
    gemm(h_bf, wt, fb2, /*res=*/x, y, nullptr, DNC_M, DNC_D, DNC_FF, 0);
    dnc_rmsnorm<<<DNC_M, 256, 0, stream>>>(y, n2, x, x_bf);
  }

  // external memory read
  tcast(wq_mem, wt, DNC_D, DNC_D);
  gemm(x_bf, wt, bq_mem, nullptr, nullptr, mq_bf, DNC_M, DNC_D, DNC_D, 0);

  for (int c = 0; c < DNC_NCH; ++c) {
    gemm(mq_bf, memK_bf + (size_t)c * DNC_CH * DNC_D, nullptr, nullptr,
         chunk, nullptr, DNC_M, DNC_CH, DNC_D, 0);
    dnc_topk_merge<<<DNC_M, 256, 0, stream>>>(chunk, salience, top_s, top_i,
                                              c * DNC_CH, c == 0 ? 1 : 0);
  }
  dnc_memread<<<DNC_M, 256, 0, stream>>>(top_s, top_i, mem_V, read_bf);

  tcast(w_read, wt, DNC_D, DNC_D);
  gemm(read_bf, wt, b_read, /*res=*/x, y, nullptr, DNC_M, DNC_D, DNC_D, 0);
  dnc_rmsnorm<<<DNC_M, 256, 0, stream>>>(y, norm_out, x, x_bf);

  // tied LM head: logits = x @ tok_embed^T  (tok_embed already [N,K])
  gemm(x_bf, tokE_bf, nullptr, nullptr, (float*)d_out, nullptr,
       DNC_M, DNC_V, DNC_D, 0);
}
